// MultiHeadAttn_54030688584081
// MI455X (gfx1250) — matrix operand order repaired and checked by reading, not compiled, not run on hardware
//
#include <hip/hip_runtime.h>
#include <hip/hip_bf16.h>

typedef __attribute__((ext_vector_type(16))) _Float16 v16h;
typedef __attribute__((ext_vector_type(8)))  _Float16 v8h;
typedef __attribute__((ext_vector_type(8)))  float    v8f;
typedef int v4i __attribute__((vector_size(4 * sizeof(int))));
typedef unsigned int u32x4 __attribute__((vector_size(16)));
typedef int i32x8 __attribute__((vector_size(32)));
typedef int i32x4 __attribute__((vector_size(16)));

#define WMMA_F32_F16(a, b, c) \
  __builtin_amdgcn_wmma_f32_16x16x32_f16(false, (a), false, (b), (short)0, (c), false, false)

// ---- CDNA5 async global->LDS path (guarded; falls back to sync copies) ----
#if defined(__has_builtin)
#if __has_builtin(__builtin_amdgcn_global_load_async_to_lds_b128) && \
    __has_builtin(__builtin_amdgcn_s_wait_asynccnt)
#define HAVE_ASYNC_LDS 1
#endif
#if __has_builtin(__builtin_amdgcn_tensor_load_to_lds) && \
    __has_builtin(__builtin_amdgcn_s_wait_tensorcnt)
#define HAVE_TDM 1
#endif
#endif
#ifndef HAVE_ASYNC_LDS
#define HAVE_ASYNC_LDS 0
#endif
#ifndef HAVE_TDM
#define HAVE_TDM 0
#endif

#define ASG __attribute__((address_space(1)))
#define ASL __attribute__((address_space(3)))

__device__ inline void async_wait() {
#if HAVE_ASYNC_LDS
  __builtin_amdgcn_s_wait_asynccnt(0);
#endif
}

#if HAVE_TDM
// Tensor Data Mover: DMA a 2D tile (16-bit elems) global -> LDS with padding.
// D# per CDNA5 ISA 8.3/8.4. tensor dims == tile dims (tiles never OOB here).
// pad_int: 0=2,1=4,2=8,3=16,4=32 DWORDs between pads; pad_amt: code = DWORDs-1.
__device__ inline void tdm_load_2d(const void* gsrc, unsigned lds_addr,
                                   unsigned tile_w, unsigned tile_h,
                                   unsigned row_stride,
                                   unsigned pad_int, unsigned pad_amt) {
  unsigned long long ga = (unsigned long long)(size_t)gsrc;
  u32x4 g0 = { 1u,                                   // count=1, user mode
               lds_addr,                             // LDS byte address
               (unsigned)(ga & 0xFFFFFFFFu),         // global_addr[31:0]
               (unsigned)(((ga >> 32) & 0x1FFFFFFu) | (2u << 30)) }; // hi + type=2
  i32x8 g1 = { (int)((1u << 16) | (1u << 20) | (pad_int << 22) | (pad_amt << 25)),
               (int)(tile_w << 16),                  // tensor_dim0[15:0] @ bits 63:48
               (int)(tile_h << 16),                  // tensor_dim1[15:0] @ bits 95:80
               (int)(tile_w << 16),                  // tile_dim0 @ bits 127:112
               (int)tile_h,                          // tile_dim1 @ bits 143:128
               (int)row_stride,                      // tensor_dim0_stride[31:0]
               0, 0 };
  i32x4 gz  = { 0, 0, 0, 0 };
  i32x8 gz8 = { 0, 0, 0, 0, 0, 0, 0, 0 };
  __builtin_amdgcn_tensor_load_to_lds(g0, g1, gz, gz, gz8, 0);
}
__device__ inline unsigned lds_addr_of(const void* p) {
  return (unsigned)(size_t)(ASL void*)p;
}
#endif

// copy 16 halfs (32B) global -> LDS
__device__ inline void copy16_h(const _Float16* __restrict__ src, _Float16* dst) {
#if HAVE_ASYNC_LDS
  __builtin_amdgcn_global_load_async_to_lds_b128((ASG v4i*)src, (ASL v4i*)dst, 0, 0);
  __builtin_amdgcn_global_load_async_to_lds_b128((ASG v4i*)(src + 8), (ASL v4i*)(dst + 8), 0, 0);
#else
  *(v8h*)dst       = *(const v8h*)src;
  *(v8h*)(dst + 8) = *(const v8h*)(src + 8);
#endif
}

// load 16 floats, convert, store 16 halfs to LDS
__device__ inline void stage16_f32(const float* __restrict__ src, _Float16* dst) {
  float4 f0 = *(const float4*)(src);
  float4 f1 = *(const float4*)(src + 4);
  float4 f2 = *(const float4*)(src + 8);
  float4 f3 = *(const float4*)(src + 12);
  v8h h0, h1;
  h0[0]=(_Float16)f0.x; h0[1]=(_Float16)f0.y; h0[2]=(_Float16)f0.z; h0[3]=(_Float16)f0.w;
  h0[4]=(_Float16)f1.x; h0[5]=(_Float16)f1.y; h0[6]=(_Float16)f1.z; h0[7]=(_Float16)f1.w;
  h1[0]=(_Float16)f2.x; h1[1]=(_Float16)f2.y; h1[2]=(_Float16)f2.z; h1[3]=(_Float16)f2.w;
  h1[4]=(_Float16)f3.x; h1[5]=(_Float16)f3.y; h1[6]=(_Float16)f3.z; h1[7]=(_Float16)f3.w;
  *(v8h*)dst       = h0;
  *(v8h*)(dst + 8) = h1;
}

__device__ inline void stage_row16(const float* __restrict__ src, _Float16* dst) {
  stage16_f32(src, dst);
}
__device__ inline void stage_row16(const _Float16* __restrict__ src, _Float16* dst) {
  copy16_h(src, dst);
}

// ---------------------------------------------------------------------------
// Fragment loaders (from LDS, f16, contiguous-K storage)
// ---------------------------------------------------------------------------
__device__ inline v16h load_a_frag(const _Float16* row0, int ldh, int lane) {
  int r  = lane & 15;
  int kb = (lane >> 4) << 3;              // 0 or 8
  const _Float16* p = row0 + r * ldh + kb;
  v8h lo = *(const v8h*)(p);              // K kb .. kb+7
  v8h hi = *(const v8h*)(p + 16);         // K kb+16 .. kb+23
  v16h f;
#pragma unroll
  for (int i = 0; i < 8; ++i) { f[i] = lo[i]; f[i + 8] = hi[i]; }
  return f;
}

__device__ inline v16h load_b_frag(const _Float16* col0, int ldh, int lane) {
  int n  = lane & 15;
  int kb = (lane >> 4) << 3;              // 0 or 8
  const _Float16* p = col0 + n * ldh + kb;
  v8h lo = *(const v8h*)(p);
  v8h hi = *(const v8h*)(p + 16);
  v16h f;
#pragma unroll
  for (int i = 0; i < 8; ++i) { f[i] = lo[i]; f[i + 8] = hi[i]; }
  return f;
}

// ---------------------------------------------------------------------------
// Weight convert + transpose: W[K][N] f32 -> Wt[N][K] f16
// ---------------------------------------------------------------------------
__global__ __launch_bounds__(256) void convert_wT(const float* __restrict__ W,
                                                  _Float16* __restrict__ Wt,
                                                  int K, int N) {
  int idx = blockIdx.x * 256 + threadIdx.x;
  if (idx >= K * N) return;
  int k = idx / N, n = idx % N;
  Wt[(size_t)n * K + k] = (_Float16)W[idx];
}

// ---------------------------------------------------------------------------
// Tiled WMMA GEMM: C[M][N] = alpha * A[M][K] x Bt[N][K]^T
// 256 thr / 8 waves, tile 128x128, BK=32, double-buffered LDS.
// B-tile staged by TDM (wave 0 issues descriptor); A staged via VALU cvt/async.
// ---------------------------------------------------------------------------
#define GBM 128
#define GBN 128
#define GBK 32
#define GLD 40   // BK + 8 halfs pad (TDM pad: 16 DW interval, 4 DW pad)

template <typename AT, typename CT>
__global__ __launch_bounds__(256) void gemm_wmma(const AT* __restrict__ A,
                                                 const _Float16* __restrict__ Bt,
                                                 CT* __restrict__ C,
                                                 int M, int N, int K, float alpha) {
  __shared__ _Float16 As[2][GBM * GLD];
  __shared__ _Float16 Bs[2][GBN * GLD];

  const int tid  = threadIdx.x;
  const int lane = tid & 31;
  const int w    = tid >> 5;      // 0..7
  const int wm   = w & 3;         // 0..3  (32-row slice)
  const int wn   = w >> 2;        // 0..1  (64-col slice)
  const int m0   = blockIdx.y * GBM;
  const int n0   = blockIdx.x * GBN;

  // staging geometry: 128 rows x 32 halfs, 2 threads/row, 16 halfs/thread
  const int sr = tid >> 1;            // 0..127
  const int sc = (tid & 1) << 4;      // 0,16

  v8f acc[2][4] = {};

  // prologue: stage k0 = 0 into buffer 0
  stage_row16(A + (size_t)(m0 + sr) * K + sc, &As[0][sr * GLD + sc]);
#if HAVE_TDM
  if (w == 0) {
    tdm_load_2d(Bt + (size_t)n0 * K, lds_addr_of(&Bs[0][0]), GBK, GBN,
                (unsigned)K, /*pad_int 16DW*/3, /*pad 4DW*/3);
    __builtin_amdgcn_s_wait_tensorcnt(0);
  }
#else
  copy16_h(Bt + (size_t)(n0 + sr) * K + sc, &Bs[0][sr * GLD + sc]);
#endif
  async_wait();
  __syncthreads();

  int cur = 0;
  for (int k0 = 0; k0 < K; k0 += GBK) {
    // prefetch next K-tile into the other buffer
    if (k0 + GBK < K) {
      stage_row16(A + (size_t)(m0 + sr) * K + k0 + GBK + sc, &As[cur ^ 1][sr * GLD + sc]);
#if HAVE_TDM
      if (w == 0)
        tdm_load_2d(Bt + (size_t)n0 * K + k0 + GBK, lds_addr_of(&Bs[cur ^ 1][0]),
                    GBK, GBN, (unsigned)K, 3, 3);
#else
      copy16_h(Bt + (size_t)(n0 + sr) * K + k0 + GBK + sc, &Bs[cur ^ 1][sr * GLD + sc]);
#endif
    }

    // compute current buffer
    const _Float16* Abase = &As[cur][(wm * 32) * GLD];
    const _Float16* Bbase = &Bs[cur][(wn * 64) * GLD];
    v16h a0 = load_a_frag(Abase,            GLD, lane);
    v16h a1 = load_a_frag(Abase + 16 * GLD, GLD, lane);
    v16h b[4];
#pragma unroll
    for (int j = 0; j < 4; ++j) b[j] = load_b_frag(Bbase + j * 16 * GLD, GLD, lane);

#pragma unroll
    for (int j = 0; j < 4; ++j) {
      acc[0][j] = WMMA_F32_F16(a0, b[j], acc[0][j]);
      acc[1][j] = WMMA_F32_F16(a1, b[j], acc[1][j]);
    }

#if HAVE_TDM
    if (w == 0) __builtin_amdgcn_s_wait_tensorcnt(0);
#endif
    async_wait();
    __syncthreads();
    cur ^= 1;
  }

  // ---- epilogue: C layout m = r + 8*(lane>=16), n = lane&15 ----
  const int moff = (lane >> 4) << 3;
  const int ncol = lane & 15;
#pragma unroll
  for (int fm = 0; fm < 2; ++fm)
#pragma unroll
    for (int fn = 0; fn < 4; ++fn)
#pragma unroll
      for (int r = 0; r < 8; ++r) {
        int m = m0 + wm * 32 + fm * 16 + moff + r;
        int n = n0 + wn * 64 + fn * 16 + ncol;
        C[(size_t)m * N + n] = (CT)(alpha * acc[fm][fn][r]);
      }
}

// ---------------------------------------------------------------------------
// Flash attention per (b, h, 64-row q tile). head_dim = 64, causal.
// K tile staged by TDM; V transposed via VALU; softmax 4 threads/row.
// ---------------------------------------------------------------------------
#define ALD 72   // 64 + 8 halfs pad (TDM pad: 32 DW interval, 4 DW pad)

__global__ __launch_bounds__(256) void flash_attn(const _Float16* __restrict__ qf,
                                                  const _Float16* __restrict__ kf,
                                                  const _Float16* __restrict__ vf,
                                                  _Float16* __restrict__ ctx) {
  const int qi = blockIdx.x;        // 0..31
  const int h  = blockIdx.y;        // 0..15
  const int b  = blockIdx.z;        // 0..3
  const int qbase = qi * 64;
  const int tid  = threadIdx.x;
  const int lane = tid & 31;
  const int w    = tid >> 5;
  const int wq   = w >> 1;          // 0..3
  const int wn   = w & 1;           // 0..1

  __shared__ _Float16 Qs[64 * ALD];
  __shared__ _Float16 Ks[64 * ALD];
  __shared__ _Float16 Vt[64 * ALD]; // transposed: Vt[d][k]
  __shared__ _Float16 Ps[64 * ALD];
  __shared__ float    Ss[64 * 64];
  __shared__ float    m_s[64], l_s[64], c_s[64];
  __shared__ float    redmax[64 * 4], redsum[64 * 4];

  const size_t rstride = 1024;
  const size_t base_q = ((size_t)b * 2048 + qbase) * rstride + (size_t)h * 64;

  const int r4 = tid >> 2;          // 0..63
  const int c4 = (tid & 3) << 4;    // 0,16,32,48

  // load Q tile (64 x 64 halfs)
  copy16_h(qf + base_q + (size_t)r4 * rstride + c4, Qs + r4 * ALD + c4);
  if (tid < 64) { m_s[tid] = -3.0e38f; l_s[tid] = 0.0f; }

  v8f o0 = {}, o1 = {};
  const int moff = (lane >> 4) << 3;
  const int ncol = lane & 15;

  for (int j = 0; j <= qi; ++j) {
    const int kb = j * 64;
    const size_t base_k = ((size_t)b * 2048 + kb) * rstride + (size_t)h * 64;

    // stage K tile (64x64, row stride 1024) and V transposed
#if HAVE_TDM
    if (w == 0)
      tdm_load_2d(kf + base_k, lds_addr_of(Ks), 64, 64, 1024,
                  /*pad_int 32DW*/4, /*pad 4DW*/3);
#else
    copy16_h(kf + base_k + (size_t)r4 * rstride + c4, Ks + r4 * ALD + c4);
#endif
    {
      const _Float16* vsrc = vf + base_k + (size_t)r4 * rstride + c4;
      v8h v0 = *(const v8h*)vsrc;
      v8h v1 = *(const v8h*)(vsrc + 8);
#pragma unroll
      for (int e = 0; e < 8; ++e) {
        Vt[(c4 + e) * ALD + r4]     = v0[e];
        Vt[(c4 + 8 + e) * ALD + r4] = v1[e];
      }
    }
    async_wait();
#if HAVE_TDM
    if (w == 0) __builtin_amdgcn_s_wait_tensorcnt(0);
#endif
    __syncthreads();

    // scores: S = Q x K^T  (rows wq*16.., cols wn*32..)
    v8f s0 = {}, s1 = {};
#pragma unroll
    for (int d0 = 0; d0 < 64; d0 += 32) {
      v16h aq  = load_a_frag(Qs + (wq * 16) * ALD + d0, ALD, lane);
      v16h bk0 = load_b_frag(Ks + (wn * 32 +  0) * ALD + d0, ALD, lane);
      v16h bk1 = load_b_frag(Ks + (wn * 32 + 16) * ALD + d0, ALD, lane);
      s0 = WMMA_F32_F16(aq, bk0, s0);
      s1 = WMMA_F32_F16(aq, bk1, s1);
    }
    // causal mask + write to LDS
#pragma unroll
    for (int r = 0; r < 8; ++r) {
      int mr  = wq * 16 + moff + r;
      int n0i = wn * 32 + ncol;
      int n1i = n0i + 16;
      float sv0 = s0[r], sv1 = s1[r];
      if (kb + n0i > qbase + mr) sv0 = -1.0e9f;
      if (kb + n1i > qbase + mr) sv1 = -1.0e9f;
      Ss[mr * 64 + n0i] = sv0;
      Ss[mr * 64 + n1i] = sv1;
    }
    __syncthreads();

    // online softmax: 4 threads per row, 16 columns each
    {
      const int row = r4;
      const int q4  = tid & 3;
      const int cb  = q4 << 4;
      float pm = -3.0e38f;
#pragma unroll
      for (int n = 0; n < 16; ++n) pm = fmaxf(pm, Ss[row * 64 + cb + n]);
      redmax[row * 4 + q4] = pm;
      __syncthreads();
      float mOld = m_s[row];
      float mNew = fmaxf(mOld,
                   fmaxf(fmaxf(redmax[row * 4 + 0], redmax[row * 4 + 1]),
                         fmaxf(redmax[row * 4 + 2], redmax[row * 4 + 3])));
      float psum = 0.0f;
#pragma unroll
      for (int n = 0; n < 16; ++n) {
        float p = __expf(Ss[row * 64 + cb + n] - mNew);
        Ps[row * ALD + cb + n] = (_Float16)p;
        psum += p;
      }
      redsum[row * 4 + q4] = psum;
      __syncthreads();
      if (q4 == 0) {
        float s4 = redsum[row * 4 + 0] + redsum[row * 4 + 1] +
                   redsum[row * 4 + 2] + redsum[row * 4 + 3];
        float corr = __expf(mOld - mNew);
        l_s[row] = l_s[row] * corr + s4;
        m_s[row] = mNew;
        c_s[row] = corr;
      }
      __syncthreads();
    }

    // rescale running accumulator
#pragma unroll
    for (int r = 0; r < 8; ++r) {
      float cfac = c_s[wq * 16 + moff + r];
      o0[r] *= cfac;
      o1[r] *= cfac;
    }
    // O += P x V
#pragma unroll
    for (int d0 = 0; d0 < 64; d0 += 32) {
      v16h ap  = load_a_frag(Ps + (wq * 16) * ALD + d0, ALD, lane);
      v16h bv0 = load_b_frag(Vt + (wn * 32 +  0) * ALD + d0, ALD, lane);
      v16h bv1 = load_b_frag(Vt + (wn * 32 + 16) * ALD + d0, ALD, lane);
      o0 = WMMA_F32_F16(ap, bv0, o0);
      o1 = WMMA_F32_F16(ap, bv1, o1);
    }
    __syncthreads();
  }

  // normalize and write ctx (f16)
#pragma unroll
  for (int r = 0; r < 8; ++r) {
    int mr = wq * 16 + moff + r;
    float inv = 1.0f / l_s[mr];
    int n0i = wn * 32 + ncol;
    ctx[base_q + (size_t)mr * rstride + n0i]      = (_Float16)(o0[r] * inv);
    ctx[base_q + (size_t)mr * rstride + n0i + 16] = (_Float16)(o1[r] * inv);
  }
}

// ---------------------------------------------------------------------------
// Host-side launcher
// ---------------------------------------------------------------------------
extern "C" void kernel_launch(void* const* d_in, const int* in_sizes, int n_in,
                              void* d_out, int out_size, void* d_ws, size_t ws_size,
                              hipStream_t stream) {
  const float* Xq = (const float*)d_in[0];
  const float* Xk = (const float*)d_in[1];
  const float* Xv = (const float*)d_in[2];
  // d_in[3] = additive causal mask buffer — causality computed analytically
  const float* Wq = (const float*)d_in[4];
  const float* Wk = (const float*)d_in[5];
  const float* Wv = (const float*)d_in[6];
  const float* Wo = (const float*)d_in[7];

  const int M = 4 * 2048;   // 8192 rows
  const int D = 1024;

  char* ws = (char*)d_ws;
  const size_t ACT = (size_t)M * D * sizeof(_Float16);     // 16 MB
  const size_t WSZ = (size_t)D * D * sizeof(_Float16);     // 2 MB
  _Float16* qf  = (_Float16*)(ws + 0 * ACT);
  _Float16* kfp = (_Float16*)(ws + 1 * ACT);
  _Float16* vfp = (_Float16*)(ws + 2 * ACT);
  _Float16* ctx = (_Float16*)(ws + 3 * ACT);
  _Float16* WqT = (_Float16*)(ws + 4 * ACT + 0 * WSZ);
  _Float16* WkT = (_Float16*)(ws + 4 * ACT + 1 * WSZ);
  _Float16* WvT = (_Float16*)(ws + 4 * ACT + 2 * WSZ);
  _Float16* WoT = (_Float16*)(ws + 4 * ACT + 3 * WSZ);

  dim3 tb(256);
  // 1) weight convert + transpose
  int wblocks = (D * D + 255) / 256;
  convert_wT<<<wblocks, tb, 0, stream>>>(Wq, WqT, D, D);
  convert_wT<<<wblocks, tb, 0, stream>>>(Wk, WkT, D, D);
  convert_wT<<<wblocks, tb, 0, stream>>>(Wv, WvT, D, D);
  convert_wT<<<wblocks, tb, 0, stream>>>(Wo, WoT, D, D);

  // 2) QKV projections (Q pre-scaled by head_dim^-0.5 = 0.125)
  dim3 ggrid(D / GBN, M / GBM);   // (8, 64)
  gemm_wmma<float, _Float16><<<ggrid, tb, 0, stream>>>(Xq, WqT, qf,  M, D, D, 0.125f);
  gemm_wmma<float, _Float16><<<ggrid, tb, 0, stream>>>(Xk, WkT, kfp, M, D, D, 1.0f);
  gemm_wmma<float, _Float16><<<ggrid, tb, 0, stream>>>(Xv, WvT, vfp, M, D, D, 1.0f);

  // 3) causal flash attention
  dim3 agrid(2048 / 64, 16, 4);   // (q tiles, heads, batch)
  flash_attn<<<agrid, tb, 0, stream>>>(qf, kfp, vfp, ctx);

  // 4) output projection -> fp32 d_out
  gemm_wmma<_Float16, float><<<ggrid, tb, 0, stream>>>(ctx, WoT, (float*)d_out, M, D, D, 1.0f);
}
